// Attention_37761352466806
// MI455X (gfx1250) — matrix operand order repaired and checked by reading, not compiled, not run on hardware
//
#include <hip/hip_runtime.h>
#include <hip/hip_bf16.h>
#include <math.h>

typedef __attribute__((ext_vector_type(16))) _Float16 v16h;
typedef __attribute__((ext_vector_type(8)))  _Float16 v8h;
typedef __attribute__((ext_vector_type(8)))  float    v8f;
typedef __attribute__((ext_vector_type(4)))  unsigned v4u;
typedef __attribute__((ext_vector_type(8)))  int      v8i;
typedef __attribute__((ext_vector_type(4)))  int      v4i;

#define EDIM 256
#define NAG  8192          // B*N = 256*32
#define NN   31            // neighbors per agent
#define PI_F 3.14159265358979f
#define LBP  40            // padded LDS row stride (halves): 16 DW tile row + 4 DW pad

#ifndef USE_TDM
#if defined(__has_builtin)
#if __has_builtin(__builtin_amdgcn_tensor_load_to_lds)
#define USE_TDM 1
#endif
#endif
#endif
#ifndef USE_TDM
#define USE_TDM 0
#endif

static __device__ inline float lrelu01(float x) { return x > 0.f ? x : 0.01f * x; }
static __device__ inline float wrap_pi(float a) {
    float t = fmodf(a + PI_F, 2.f * PI_F);
    if (t < 0.f) t += 2.f * PI_F;
    return t - PI_F;
}
static __device__ inline unsigned hashu(unsigned x) {
    x ^= x >> 16; x *= 0x7feb352dU; x ^= x >> 15; x *= 0x846ca68bU; x ^= x >> 16;
    return x;
}
static __device__ inline v8f wmma_f16(v16h a, v16h b, v8f c) {
    return __builtin_amdgcn_wmma_f32_16x16x32_f16(false, a, false, b, (short)0, c, false, false);
}
static __device__ inline v16h cat8(const _Float16* lo, const _Float16* hi) {
    return __builtin_shufflevector(*(const v8h*)lo, *(const v8h*)hi,
                                   0, 1, 2, 3, 4, 5, 6, 7, 8, 9, 10, 11, 12, 13, 14, 15);
}
static __device__ inline unsigned lds_addr32(const void* p) {
    return (unsigned)(unsigned long long)p;   // LDS aperture: addr[31:0] = LDS byte offset
}

#if USE_TDM
// TDM: load a (rows x 32-halves) f16 tile (row stride rowLen halves in memory) into LDS,
// padding 4 DWORDs after every 16 DWORDs -> LDS row stride = LBP(40) halves.
static __device__ inline void tdm_load_tile(const _Float16* src, unsigned ldsAddr,
                                            int rowLen, int rows) {
    unsigned long long ga = (unsigned long long)(const void*)src;
    v4u g0;
    g0.x = 1u;                                                 // count=1, user D#
    g0.y = ldsAddr;                                            // lds_addr (bytes)
    g0.z = (unsigned)ga;                                       // global_addr[31:0]
    g0.w = ((unsigned)(ga >> 32) & 0x01FFFFFFu) | (2u << 30);  // global_addr[56:32], type=2
    v8i g1;
    g1[0] = (1 << 16) | (1 << 20) | (3 << 22) | (3 << 25);     // data_size=2B, pad_en, 16DW, +4DW
    g1[1] = (rowLen & 0xFFFF) << 16;                           // [63:48] = tensor_dim0 lo16
    g1[2] = ((rowLen >> 16) & 0xFFFF) | ((rows & 0xFFFF) << 16); // dim0 hi16 | tensor_dim1 lo16
    g1[3] = ((rows >> 16) & 0xFFFF) | (32 << 16);              // dim1 hi16 | tile_dim0 = 32
    g1[4] = (rows & 0xFFFF);                                   // tile_dim1 = rows, tile_dim2 = 0
    g1[5] = rowLen;                                            // tensor_dim0_stride[31:0]
    g1[6] = 0;                                                 // stride hi16 | dim1_stride lo16
    g1[7] = 0;
    v4i gz4 = {0, 0, 0, 0};
    v8i gz8 = {0, 0, 0, 0, 0, 0, 0, 0};
    __builtin_amdgcn_tensor_load_to_lds(g0, g1, gz4, gz4, gz8, 0);
}
#endif

// ---------------------------------------------------------------- f32 KxN -> f16 NxK (transpose)
__global__ __launch_bounds__(256) void cvt_t_kernel(const float* __restrict__ src,
                                                    _Float16* __restrict__ dst, int K, int N) {
    __shared__ float tile[32][33];
    int k0 = blockIdx.y * 32, n0 = blockIdx.x * 32;
    int tx = threadIdx.x & 31, ty = threadIdx.x >> 5;   // 32 x 8
#pragma unroll
    for (int r = ty; r < 32; r += 8)
        tile[r][tx] = src[(size_t)(k0 + r) * N + n0 + tx];
    __syncthreads();
#pragma unroll
    for (int r = ty; r < 32; r += 8)
        dst[(size_t)(n0 + r) * K + k0 + tx] = (_Float16)tile[tx][r];
}

// ---------------------------------------------------------------- agent embed MLP (5->128->256)
__global__ __launch_bounds__(128) void agent_kernel(
    const float* __restrict__ emb, const float* __restrict__ e1w, const float* __restrict__ e1b,
    const float* __restrict__ e2w, const float* __restrict__ e2b, _Float16* __restrict__ agent16) {
    __shared__ float sE[5];
    __shared__ float sH[128];
    int a = blockIdx.x, tid = threadIdx.x;
    if (tid < 5) sE[tid] = emb[a * 11 + 4 + tid];
    __syncthreads();
    {
        float h = e1b[tid];
#pragma unroll
        for (int t = 0; t < 5; ++t) h += sE[t] * e1w[t * 128 + tid];
        sH[tid] = lrelu01(h);
    }
    __syncthreads();
    for (int c = tid; c < EDIM; c += 128) {
        float o = e2b[c];
        for (int t = 0; t < 128; ++t) o += sH[t] * e2w[t * EDIM + c];
        agent16[(size_t)a * EDIM + c] = (_Float16)lrelu01(o);
    }
}

// ---------------------------------------------------------------- generic WMMA GEMM, 128x64 tile
// C = act(A(f16, MxK) @ B + bias), BT is f16 N x K row-major. ACT: 0=none 1=relu 2=lrelu
template <int ACT, bool HASBIAS, bool WF, bool WH>
__global__ __launch_bounds__(256) void gemm_kernel(
    const _Float16* __restrict__ A, const _Float16* __restrict__ BT, const float* __restrict__ bias,
    float* __restrict__ Cf, _Float16* __restrict__ Ch, int M, int N, int K) {
#if USE_TDM
    __shared__ __align__(16) _Float16 As[2][128 * LBP];
    __shared__ __align__(16) _Float16 Bs[2][64 * LBP];
#else
    __shared__ __align__(16) _Float16 As[1][128 * LBP];
    __shared__ __align__(16) _Float16 Bs[1][64 * LBP];
#endif
    int tid = threadIdx.x, wave = tid >> 5, lane = tid & 31;
    int m0 = blockIdx.y * 128, n0 = blockIdx.x * 64;
    v8f z = {0, 0, 0, 0, 0, 0, 0, 0};
    v8f acc[2][2] = {{z, z}, {z, z}};
    int k0 = (lane >> 4) * 8;
    int kb = (lane >> 4) * 16;
    int mr0 = (wave & 3) * 32 + (lane & 15);   // local row of first m-tile
    int nc = (wave >> 2) * 32 + (lane & 15);
    int nk = K >> 5;
#if USE_TDM
    if (wave == 0) {
        tdm_load_tile(A + (size_t)m0 * K, lds_addr32(&As[0][0]), K, 128);
        tdm_load_tile(BT + (size_t)n0 * K, lds_addr32(&Bs[0][0]), K, 64);
    }
#endif
    for (int i = 0; i < nk; ++i) {
        int kk = i << 5;
#if USE_TDM
        if (wave == 0) {
            if (i + 1 < nk) {
                tdm_load_tile(A + (size_t)m0 * K + kk + 32, lds_addr32(&As[(i + 1) & 1][0]), K, 128);
                tdm_load_tile(BT + (size_t)n0 * K + kk + 32, lds_addr32(&Bs[(i + 1) & 1][0]), K, 64);
                __builtin_amdgcn_s_wait_tensorcnt(2);
            } else {
                __builtin_amdgcn_s_wait_tensorcnt(0);
            }
        }
        const _Float16* as = &As[i & 1][0];
        const _Float16* bs = &Bs[i & 1][0];
#else
#pragma unroll
        for (int ss = 0; ss < 2; ++ss) {       // stage A: 128 rows x 32 k
            int e = (tid + ss * 256) * 8;
            int r = e >> 5, c = e & 31;
            *(uint4*)&As[0][r * LBP + c] = *(const uint4*)&A[(size_t)(m0 + r) * K + kk + c];
        }
        { int r = tid >> 2, c = (tid & 3) * 8; // stage BT: 64 n-rows x 32 k
          *(uint4*)&Bs[0][r * LBP + c] = *(const uint4*)&BT[(size_t)(n0 + r) * K + kk + c]; }
        const _Float16* as = &As[0][0];
        const _Float16* bs = &Bs[0][0];
#endif
        __syncthreads();
        v16h a0 = cat8(&as[mr0 * LBP + k0], &as[mr0 * LBP + 16 + k0]);
        v16h a1 = cat8(&as[(mr0 + 16) * LBP + k0], &as[(mr0 + 16) * LBP + 16 + k0]);
        v16h b0 = cat8(&bs[nc * LBP + k0], &bs[nc * LBP + 16 + k0]);
        v16h b1 = cat8(&bs[(nc + 16) * LBP + k0], &bs[(nc + 16) * LBP + 16 + k0]);
        acc[0][0] = wmma_f16(a0, b0, acc[0][0]);
        acc[0][1] = wmma_f16(a0, b1, acc[0][1]);
        acc[1][0] = wmma_f16(a1, b0, acc[1][0]);
        acc[1][1] = wmma_f16(a1, b1, acc[1][1]);
        __syncthreads();
    }
    int col0 = n0 + (wave >> 2) * 32 + (lane & 15);
#pragma unroll
    for (int mi = 0; mi < 2; ++mi) {
        int mrow = m0 + (wave & 3) * 32 + mi * 16 + (lane >= 16 ? 8 : 0);
#pragma unroll
        for (int r = 0; r < 8; ++r) {
            float v0 = acc[mi][0][r], v1 = acc[mi][1][r];
            if (HASBIAS) { v0 += bias[col0]; v1 += bias[col0 + 16]; }
            if (ACT == 1) { v0 = fmaxf(v0, 0.f); v1 = fmaxf(v1, 0.f); }
            if (ACT == 2) { v0 = lrelu01(v0); v1 = lrelu01(v1); }
            int row = mrow + r;
            if (WF) { Cf[(size_t)row * N + col0] = v0; Cf[(size_t)row * N + col0 + 16] = v1; }
            if (WH) { Ch[(size_t)row * N + col0] = (_Float16)v0; Ch[(size_t)row * N + col0 + 16] = (_Float16)v1; }
        }
    }
}

// ---------------------------------------------------------------- in-block 64x256 @ 256x256 WMMA
// MODE 0: dual dot of raw acc with sHe (per-column pair) -> sPart[0/1][row][ng]
// MODE 1: dot of relu(acc + sRow[la*256+col]) with sAs2 -> sPart[0][row][ng]
template <int MODE>
static __device__ inline void gemm64_dot(const _Float16* sA, const _Float16* __restrict__ BT,
                                         _Float16* sB, const _Float16* sHe, const _Float16* sAs2,
                                         const float* sRow, float* sPart, int tid) {
    int wave = tid >> 5, lane = tid & 31;
    int mt0 = (wave & 1) * 2;       // m-tiles {mt0, mt0+1}; rows mt0*16 .. mt0*16+31 (agent = wave&1)
    int ng  = wave >> 1;            // n-group: columns ng*64 .. ng*64+63
    int la  = wave & 1;
    v8f z = {0, 0, 0, 0, 0, 0, 0, 0};
    v8f acc[2][4] = {{z, z, z, z}, {z, z, z, z}};
    int k0 = (lane >> 4) * 8;
    int kb = (lane >> 4) * 16;
    int mrow0 = mt0 * 16 + (lane & 15);
    int mrow1 = mrow0 + 16;
#if USE_TDM
    unsigned ldsB = lds_addr32(sB);
#endif
#pragma unroll
    for (int kk = 0; kk < 256; kk += 32) {
#if USE_TDM
        if (wave == 0) {
            tdm_load_tile(BT + kk, ldsB, 256, 256);
            __builtin_amdgcn_s_wait_tensorcnt(0);
        }
#else
#pragma unroll
        for (int ss = 0; ss < 4; ++ss) {       // stage BT tile: 256 n-rows x 32 k
            int e = (tid + ss * 256) * 8;
            int r = e >> 5, c = e & 31;
            *(uint4*)&sB[r * LBP + c] = *(const uint4*)&BT[(size_t)r * 256 + kk + c];
        }
#endif
        __syncthreads();
        v16h a0 = cat8(&sA[mrow0 * 256 + kk + k0], &sA[mrow0 * 256 + kk + 16 + k0]);
        v16h a1 = cat8(&sA[mrow1 * 256 + kk + k0], &sA[mrow1 * 256 + kk + 16 + k0]);
#pragma unroll
        for (int ni = 0; ni < 4; ++ni) {
            int nc = ng * 64 + ni * 16 + (lane & 15);
            v16h bf = cat8(&sB[nc * LBP + k0], &sB[nc * LBP + 16 + k0]);
            acc[0][ni] = wmma_f16(a0, bf, acc[0][ni]);
            acc[1][ni] = wmma_f16(a1, bf, acc[1][ni]);
        }
        __syncthreads();
    }
    // register-space row-dot epilogue
#pragma unroll
    for (int mi = 0; mi < 2; ++mi) {
        float p0[8] = {0, 0, 0, 0, 0, 0, 0, 0};
        float p1[8] = {0, 0, 0, 0, 0, 0, 0, 0};
#pragma unroll
        for (int ni = 0; ni < 4; ++ni) {
            int col = ng * 64 + ni * 16 + (lane & 15);
            float w0 = (MODE == 0) ? (float)sHe[col * 2] : (float)sAs2[col];
            float w1 = (MODE == 0) ? (float)sHe[col * 2 + 1] : 0.f;
            float ar = (MODE == 1) ? sRow[la * 256 + col] : 0.f;
#pragma unroll
            for (int r = 0; r < 8; ++r) {
                float v = acc[mi][ni][r];
                if (MODE == 1) v = fmaxf(v + ar, 0.f);
                p0[r] += v * w0;
                if (MODE == 0) p1[r] += v * w1;
            }
        }
#pragma unroll
        for (int mask = 1; mask <= 8; mask <<= 1) {
#pragma unroll
            for (int r = 0; r < 8; ++r) {
                p0[r] += __shfl_xor(p0[r], mask, 32);
                if (MODE == 0) p1[r] += __shfl_xor(p1[r], mask, 32);
            }
        }
        if ((lane & 15) == 0) {
            int rbase = (mt0 + mi) * 16 + (lane >> 4) * 8;
#pragma unroll
            for (int r = 0; r < 8; ++r) {
                sPart[(rbase + r) * 4 + ng] = p0[r];
                if (MODE == 0) sPart[(64 + rbase + r) * 4 + ng] = p1[r];
            }
        }
    }
}

// ---------------------------------------------------------------- fused per-agent edge kernel
// One block handles 2 agents (64 padded edge rows).
__global__ __launch_bounds__(256) void edge_kernel(
    const float* __restrict__ emb,
    const float* __restrict__ s1,        // agent_embed @ hm1_w[:256] + hm1_b  (f32, 8192x256)
    const float* __restrict__ a1,        // q @ as1_w[:256] + as1_b            (f32, 8192x256)
    const float* __restrict__ hm1w,      // f32 263x256
    const float* __restrict__ hm2b,
    const float* __restrict__ hew, const float* __restrict__ heb,
    const float* __restrict__ kw,  const float* __restrict__ vw, const float* __restrict__ vb,
    const float* __restrict__ as2w, const float* __restrict__ as2b,
    const _Float16* __restrict__ hm2T,      // f16 256x256 (N x K)
    const _Float16* __restrict__ as1hiT,    // f16 256x256 (N x K), rows 256..511 of as1_w
    const _Float16* __restrict__ agent16,
    _Float16* __restrict__ concat16,        // 8192x512
    float* __restrict__ entws,
    float* __restrict__ out_logits, float* __restrict__ out_dist,
    float* __restrict__ out_hw, float* __restrict__ out_comb) {
    __shared__ __align__(16) _Float16 sA[64 * 256];     // 32 KB
    __shared__ __align__(16) _Float16 sB[256 * LBP];    // 20 KB
    __shared__ float sEdge[64 * 12];                    // 3 KB
    __shared__ _Float16 sHe[512];
    __shared__ _Float16 sAs2[256];
    __shared__ float sRow[2 * 256];                     // s1 rows, later a1 rows
    __shared__ float sPart[2 * 64 * 4];
    __shared__ float sL0[64], sL1[64], sHW[64], sScore[64], sComb[64];
    __shared__ float sPi[2][6];
    __shared__ float sBdot[2];

    int a0 = blockIdx.x * 2;
    int tid = threadIdx.x;

    sHe[tid] = (_Float16)hew[tid];
    sHe[tid + 256] = (_Float16)hew[tid + 256];
    sAs2[tid] = (_Float16)as2w[tid];
    sRow[tid] = s1[(size_t)a0 * EDIM + tid];
    sRow[256 + tid] = s1[(size_t)(a0 + 1) * EDIM + tid];
    if (tid < 2) {
        const float* e = emb + (size_t)(a0 + tid) * 11;
        sPi[tid][0] = e[0]; sPi[tid][1] = e[1]; sPi[tid][2] = e[2]; sPi[tid][3] = e[3];
        sPi[tid][4] = atan2f(e[3], e[2]);
    }
    if (tid == 2) {   // fold hm2_b into the he-dot: constant per logit component
        float b0 = 0.f, b1 = 0.f;
        for (int c = 0; c < EDIM; ++c) { b0 += hm2b[c] * hew[c * 2]; b1 += hm2b[c] * hew[c * 2 + 1]; }
        sBdot[0] = b0; sBdot[1] = b1;
    }
    __syncthreads();

    // ---- edge features (10 per neighbor), 2 agents ----
    if (tid < 62) {
        int la = tid >= NN ? 1 : 0;
        int jj = tid - la * NN;
        int aa = a0 + la;
        int i = aa & 31, b = aa >> 5;
        int j = jj + (jj >= i ? 1 : 0);
        const float* ej = emb + (size_t)(b * 32 + j) * 11;
        float dx = ej[0] - sPi[la][0], dy = ej[1] - sPi[la][1];
        float dist = sqrtf(dx * dx + dy * dy);
        float ang = wrap_pi(atan2f(dy, dx) - sPi[la][4]);
        float gx = ej[9] - sPi[la][0], gy = ej[10] - sPi[la][1];
        float gd = sqrtf(gx * gx + gy * gy);
        float ga = wrap_pi(atan2f(gy, gx) - sPi[la][4]);
        float* e10 = &sEdge[(la * 32 + jj) * 12];
        e10[0] = dist / 12.f; e10[1] = cosf(ang); e10[2] = sinf(ang);
        e10[3] = ej[2]; e10[4] = ej[3]; e10[5] = ej[7]; e10[6] = ej[8];
        e10[7] = gd; e10[8] = cosf(ga); e10[9] = sinf(ga);
        out_dist[(size_t)aa * NN + jj] = dist;
    } else if (tid < 64) {
        int la = tid - 62;
        for (int q = 0; q < 12; ++q) sEdge[(la * 32 + 31) * 12 + q] = 0.f;
    }
    __syncthreads();

    // ---- X1 = relu(s1_shared + edge7 @ hm1_w[256:263]) -> sA (f16, 64 rows) ----
    {
        int r = tid >> 2, c0 = (tid & 3) * 64;
        int la = r >> 5, rr = r & 31;
        const float* e10 = &sEdge[r * 12];
        for (int c = c0; c < c0 + 64; ++c) {
            float v = 0.f;
            if (rr < NN) {
                v = sRow[la * 256 + c];
#pragma unroll
                for (int q = 0; q < 7; ++q) v += e10[q] * hm1w[(256 + q) * EDIM + c];
                v = fmaxf(v, 0.f);
            }
            sA[r * 256 + c] = (_Float16)v;
        }
    }
    __syncthreads();

    // ---- H2 = X1 @ hm2_w, fused he-dot epilogue ----
    gemm64_dot<0>(sA, hm2T, sB, sHe, sAs2, sRow, sPart, tid);
    __syncthreads();

    // ---- logits finalize ----
    if (tid < 128) {
        int r = tid >> 1, cp = tid & 1;
        const float* pp = &sPart[(cp * 64 + r) * 4];
        float s = pp[0] + pp[1] + pp[2] + pp[3] + sBdot[cp] + heb[cp];
        if (cp == 0) sL0[r] = s;
        else {
            sL1[r] = s;
            if ((r & 31) < NN) out_logits[(size_t)(a0 + (r >> 5)) * NN + (r & 31)] = s;
        }
    }
    __syncthreads();

    // ---- gumbel-softmax hard weights + swap sRow to a1 rows ----
    if (tid < 64 && (tid & 31) < NN) {
        int aa = a0 + (tid >> 5), jj = tid & 31;
        unsigned base = (unsigned)(aa * 62 + jj * 2);
        float u0 = (hashu(base)     >> 8) * (1.f / 16777216.f);
        float u1 = (hashu(base + 1) >> 8) * (1.f / 16777216.f);
        u0 = fminf(fmaxf(u0, 1e-6f), 1.f - 1e-6f);
        u1 = fminf(fmaxf(u1, 1e-6f), 1.f - 1e-6f);
        float g0 = -logf(-logf(u0)), g1 = -logf(-logf(u1));
        float z0 = (sL0[tid] + g0) * 2.f;    // /TAU, TAU=0.5
        float z1 = (sL1[tid] + g1) * 2.f;
        float m = fmaxf(z0, z1);
        float e0 = __expf(z0 - m), e1 = __expf(z1 - m);
        float p1 = e1 / (e0 + e1);
        sHW[tid] = p1;
        out_hw[(size_t)aa * NN + jj] = p1;
    }
    sRow[tid] = a1[(size_t)a0 * EDIM + tid];
    sRow[256 + tid] = a1[(size_t)(a0 + 1) * EDIM + tid];
    __syncthreads();

    // ---- k_feat = lrelu(edge10 @ k_w) -> sA ----
    {
        int r = tid >> 2, c0 = (tid & 3) * 64;
        int rr = r & 31;
        const float* e10 = &sEdge[r * 12];
        for (int c = c0; c < c0 + 64; ++c) {
            float v = 0.f;
            if (rr < NN) {
#pragma unroll
                for (int q = 0; q < 10; ++q) v += e10[q] * kw[q * EDIM + c];
                v = lrelu01(v);
            }
            sA[r * 256 + c] = (_Float16)v;
        }
    }
    __syncthreads();

    // ---- A1 = relu(a1_shared + k_feat @ as1_w[256:512]), fused as2-dot ----
    gemm64_dot<1>(sA, as1hiT, sB, sHe, sAs2, sRow, sPart, tid);
    __syncthreads();

    if (tid < 64) {
        const float* pp = &sPart[tid * 4];
        sScore[tid] = pp[0] + pp[1] + pp[2] + pp[3] + as2b[0];
    }
    __syncthreads();

    // ---- masked softmax / combined / entropy (one thread per agent) ----
    if (tid < 2) {
        int la = tid, base = la * 32;
        float comb[NN];
        float msum = 0.f;
        for (int j = 0; j < NN; ++j) { comb[j] = sHW[base + j] > 0.5f ? 1.f : 0.f; msum += comb[j]; }
        if (msum < 1e-6f) {
            for (int j = 0; j < NN; ++j) comb[j] = 0.f;
        } else {
            float mx = -1e30f;
            for (int j = 0; j < NN; ++j) if (comb[j] > 0.f) mx = fmaxf(mx, sScore[base + j]);
            float se = 0.f;
            for (int j = 0; j < NN; ++j) {
                comb[j] = comb[j] > 0.f ? __expf(sScore[base + j] - mx) : 0.f;
                se += comb[j];
            }
            float inv = 1.f / se;
            for (int j = 0; j < NN; ++j) comb[j] *= inv;
        }
        float cs = 0.f;
        for (int j = 0; j < NN; ++j) cs += comb[j];
        float invc = 1.f / (cs + 1e-6f), ent = 0.f;
        for (int j = 0; j < NN; ++j) {
            float cw = comb[j] * invc;
            ent -= cw * logf(cw + 1e-6f);
        }
        entws[a0 + la] = ent;
        for (int j = 0; j < NN; ++j) {
            sComb[base + j] = comb[j];
            out_comb[(size_t)(a0 + la) * NN + j] = comb[j];
        }
    }
    __syncthreads();

    // ---- attn_out (recompute v_j on the fly) + concat for decoder ----
    {
        int c = tid;
        float wv[10], vbc = vb[c];
#pragma unroll
        for (int q = 0; q < 10; ++q) wv[q] = vw[q * EDIM + c];
#pragma unroll
        for (int la = 0; la < 2; ++la) {
            float ao = 0.f;
            for (int j = 0; j < NN; ++j) {
                const float* e10 = &sEdge[(la * 32 + j) * 12];
                float v = vbc;
#pragma unroll
                for (int q = 0; q < 10; ++q) v += e10[q] * wv[q];
                ao += sComb[la * 32 + j] * lrelu01(v);
            }
            concat16[(size_t)(a0 + la) * 512 + c] = agent16[(size_t)(a0 + la) * EDIM + c];
            concat16[(size_t)(a0 + la) * 512 + EDIM + c] = (_Float16)ao;
        }
    }
}

// ---------------------------------------------------------------- entropy mean
__global__ __launch_bounds__(256) void ent_reduce_kernel(const float* __restrict__ e, float* __restrict__ dst) {
    __shared__ float s[256];
    float v = 0.f;
    for (int i = threadIdx.x; i < NAG; i += 256) v += e[i];
    s[threadIdx.x] = v;
    __syncthreads();
    for (int st = 128; st > 0; st >>= 1) {
        if (threadIdx.x < st) s[threadIdx.x] += s[threadIdx.x + st];
        __syncthreads();
    }
    if (threadIdx.x == 0) dst[0] = s[0] / (float)NAG;
}

// ================================================================ host
extern "C" void kernel_launch(void* const* d_in, const int* in_sizes, int n_in,
                              void* d_out, int out_size, void* d_ws, size_t ws_size,
                              hipStream_t stream) {
    const float* emb  = (const float*)d_in[0];
    const float* e1w  = (const float*)d_in[1];
    const float* e1b  = (const float*)d_in[2];
    const float* e2w  = (const float*)d_in[3];
    const float* e2b  = (const float*)d_in[4];
    const float* hm1w = (const float*)d_in[5];
    const float* hm1b = (const float*)d_in[6];
    const float* hm2w = (const float*)d_in[7];
    const float* hm2b = (const float*)d_in[8];
    const float* hew  = (const float*)d_in[9];
    const float* heb  = (const float*)d_in[10];
    const float* qw   = (const float*)d_in[11];
    const float* kw   = (const float*)d_in[12];
    const float* vw   = (const float*)d_in[13];
    const float* vb   = (const float*)d_in[14];
    const float* as1w = (const float*)d_in[15];
    const float* as1b = (const float*)d_in[16];
    const float* as2w = (const float*)d_in[17];
    const float* as2b = (const float*)d_in[18];
    const float* d1w  = (const float*)d_in[19];
    const float* d1b  = (const float*)d_in[20];
    const float* d2w  = (const float*)d_in[21];
    const float* d2b  = (const float*)d_in[22];
    float* out = (float*)d_out;

    // output layout (floats)
    const size_t O_ATT  = 0;                       // 8192*512
    const size_t O_LOG  = O_ATT + (size_t)NAG * 512;
    const size_t O_DIST = O_LOG + (size_t)NAG * NN;
    const size_t O_ENT  = O_DIST + (size_t)NAG * NN;
    const size_t O_HW   = O_ENT + 1;
    const size_t O_COMB = O_HW + (size_t)NAG * NN;

    // workspace layout
    char* ws = (char*)d_ws;
    size_t off = 0;
    auto alloc = [&](size_t bytes) { size_t o = off; off = (off + bytes + 255) & ~(size_t)255; return o; };
    _Float16* agent16  = (_Float16*)(ws + alloc((size_t)NAG * EDIM * 2));
    _Float16* q16      = (_Float16*)(ws + alloc((size_t)NAG * EDIM * 2));
    float*    s1       = (float*)   (ws + alloc((size_t)NAG * EDIM * 4));
    float*    a1       = (float*)   (ws + alloc((size_t)NAG * EDIM * 4));
    _Float16* concat16 = (_Float16*)(ws + alloc((size_t)NAG * 512 * 2));
    _Float16* dec116   = (_Float16*)(ws + alloc((size_t)NAG * 512 * 2));
    float*    entws    = (float*)   (ws + alloc((size_t)NAG * 4));
    _Float16* hm1T     = (_Float16*)(ws + alloc((size_t)65536 * 2));   // 256x256 NxK
    _Float16* hm2T     = (_Float16*)(ws + alloc((size_t)65536 * 2));
    _Float16* qwT      = (_Float16*)(ws + alloc((size_t)65536 * 2));
    _Float16* as1loT   = (_Float16*)(ws + alloc((size_t)65536 * 2));
    _Float16* as1hiT   = (_Float16*)(ws + alloc((size_t)65536 * 2));
    _Float16* d1T      = (_Float16*)(ws + alloc((size_t)262144 * 2)); // 512x512 NxK
    _Float16* d2T      = (_Float16*)(ws + alloc((size_t)262144 * 2));

    // 1) weight transpose + f32->f16 conversions (B^T layout, N x K)
    dim3 t256(8, 8), t512(16, 16);
    cvt_t_kernel<<<t256, 256, 0, stream>>>(hm1w,              hm1T,   256, 256);  // rows 0..255
    cvt_t_kernel<<<t256, 256, 0, stream>>>(hm2w,              hm2T,   256, 256);
    cvt_t_kernel<<<t256, 256, 0, stream>>>(qw,                qwT,    256, 256);
    cvt_t_kernel<<<t256, 256, 0, stream>>>(as1w,              as1loT, 256, 256);  // rows 0..255
    cvt_t_kernel<<<t256, 256, 0, stream>>>(as1w + 256 * 256,  as1hiT, 256, 256);  // rows 256..511
    cvt_t_kernel<<<t512, 256, 0, stream>>>(d1w,               d1T,    512, 512);
    cvt_t_kernel<<<t512, 256, 0, stream>>>(d2w,               d2T,    512, 512);

    // 2) agent embed
    agent_kernel<<<NAG, 128, 0, stream>>>(emb, e1w, e1b, e2w, e2b, agent16);

    // 3) shared-part GEMMs (WMMA, 128x64 tiles, TDM-staged when available)
    dim3 gA(EDIM / 64, NAG / 128);   // (4,64)
    gemm_kernel<0, false, false, true><<<gA, 256, 0, stream>>>(
        agent16, qwT, nullptr, nullptr, q16, NAG, EDIM, EDIM);                 // q (f16)
    gemm_kernel<0, true, true, false><<<gA, 256, 0, stream>>>(
        agent16, hm1T, hm1b, s1, nullptr, NAG, EDIM, EDIM);                    // s1_shared (f32)
    gemm_kernel<0, true, true, false><<<gA, 256, 0, stream>>>(
        q16, as1loT, as1b, a1, nullptr, NAG, EDIM, EDIM);                      // a1_shared (f32)

    // 4) fused per-agent edge pipeline (2 agents/block, 2x 64x256x256 WMMA GEMMs per block)
    edge_kernel<<<NAG / 2, 256, 0, stream>>>(emb, s1, a1, hm1w, hm2b, hew, heb,
                                             kw, vw, vb, as2w, as2b,
                                             hm2T, as1hiT, agent16, concat16,
                                             entws, out + O_LOG, out + O_DIST,
                                             out + O_HW, out + O_COMB);

    // 5) entropy mean
    ent_reduce_kernel<<<1, 256, 0, stream>>>(entws, out + O_ENT);

    // 6) decoder GEMMs (WMMA, leaky-relu epilogue)
    dim3 gD(512 / 64, NAG / 128);   // (8,64)
    gemm_kernel<2, true, false, true><<<gD, 256, 0, stream>>>(
        concat16, d1T, d1b, nullptr, dec116, NAG, 512, 512);
    gemm_kernel<2, true, true, false><<<gD, 256, 0, stream>>>(
        dec116, d2T, d2b, out + O_ATT, nullptr, NAG, 512, 512);
}
